// LocalToMeAttentionBlock_20160576487784
// MI455X (gfx1250) — compile-verified, matrix operand order checked by reading, not run on hardware
//
#include <hip/hip_runtime.h>
#include <hip/hip_bf16.h>
#include <math.h>

#define BB 4
#define SS 2048
#define DD 1024
#define HH 16
#define DHD 64
#define FFD 4096
#define WINS 16
#define GMD 64
#define RR 512
#define NW (SS / WINS)
#define SKEEP (SS - RR)
#define NPAIR (SS / 2)

typedef __attribute__((ext_vector_type(16))) __bf16 v16bf;
typedef __attribute__((ext_vector_type(8)))  float v8f;

union FragBF { unsigned int u[8]; v16bf v; };
union FragF  { float f[8]; v8f v; };

__device__ __forceinline__ unsigned short f2bf(float f) {
  unsigned int u = __float_as_uint(f);
  u += 0x7fffu + ((u >> 16) & 1u);
  return (unsigned short)(u >> 16);
}

// Low 32 bits of a generic pointer to __shared__ == LDS byte offset
// (LDS aperture keeps the offset in addr[31:0], ISA 10.2).
__device__ __forceinline__ unsigned lds_off(const void* p) {
  return (unsigned)(unsigned long long)p;
}

// Async copy of 16 bytes global -> LDS (GLOBAL_LOAD_ASYNC_TO_LDS_B128, ASYNCcnt).
__device__ __forceinline__ void async_copy16(unsigned ldsDst, const void* gSrc) {
  asm volatile("global_load_async_to_lds_b128 %0, %1, off"
               :: "v"(ldsDst), "v"(gSrc) : "memory");
}
__device__ __forceinline__ void wait_async0() {
  asm volatile("s_wait_asynccnt 0x0" ::: "memory");
}
__device__ __forceinline__ void wait_async1() {
  asm volatile("s_wait_asynccnt 0x1" ::: "memory");
}
__device__ __forceinline__ void wait_async2() {
  asm volatile("s_wait_asynccnt 0x2" ::: "memory");
}

// ---------------------------------------------------------------- elementwise
__global__ void k_span_add(const float* __restrict__ x, const int* __restrict__ span,
                           const unsigned char* __restrict__ mask,
                           const float* __restrict__ w_span, float* __restrict__ X) {
  size_t i = (size_t)blockIdx.x * blockDim.x + threadIdx.x;
  if (i >= (size_t)BB * SS * DD) return;
  int d = (int)(i % DD);
  size_t bs = i / DD;
  float se = log1pf((float)span[bs]) * w_span[d];
  X[i] = x[i] + (mask[bs] ? se : 0.0f);
}

__global__ void k_rms_bf16(const float* __restrict__ X, const float* __restrict__ w,
                           unsigned short* __restrict__ out) {
  __shared__ float red[256];
  int row = blockIdx.x;
  const float* xr = X + (size_t)row * DD;
  float ss = 0.f;
  for (int i = threadIdx.x; i < DD; i += 256) { float v = xr[i]; ss += v * v; }
  red[threadIdx.x] = ss; __syncthreads();
  for (int s = 128; s > 0; s >>= 1) {
    if ((int)threadIdx.x < s) red[threadIdx.x] += red[threadIdx.x + s];
    __syncthreads();
  }
  float scale = rsqrtf(red[0] / (float)DD + 1e-6f);
  for (int i = threadIdx.x; i < DD; i += 256)
    out[(size_t)row * DD + i] = f2bf(xr[i] * scale * w[i]);
}

__global__ void k_cast_bf16(const float* __restrict__ in, unsigned short* __restrict__ out, int n) {
  int i = blockIdx.x * blockDim.x + threadIdx.x;
  if (i < n) out[i] = f2bf(in[i]);
}

// Transpose-cast: out[n*K + k] = bf16(in[k*N + n]); 32x32 LDS tiles, 256 threads.
__global__ void k_castT_bf16(const float* __restrict__ in, unsigned short* __restrict__ out,
                             int K, int N) {
  __shared__ float tile[32][33];
  int nb = blockIdx.x * 32, kb = blockIdx.y * 32;
  int tx = threadIdx.x & 31, ty = threadIdx.x >> 5;   // ty: 0..7
  for (int r = ty; r < 32; r += 8)
    tile[r][tx] = in[(size_t)(kb + r) * N + nb + tx];
  __syncthreads();
  for (int r = ty; r < 32; r += 8)
    out[(size_t)(nb + r) * K + kb + tx] = f2bf(tile[tx][r]);
}

// ---------------------------------------------------------------- WMMA GEMM
// C[M,N](f32) = A[M,K](bf16,row-major) @ B[K,N] given as Bt[N,K](bf16,row-major)
// 512 threads = 16 waves; block tile 64(M) x 64(N); k-step 32.
// Double-buffered LDS tiles staged with GLOBAL_LOAD_ASYNC_TO_LDS_B128;
// next tile's DMA overlaps current tile's WMMA (s_wait_asynccnt 0x1).
__global__ void k_gemm_bf16(const unsigned short* __restrict__ A,
                            const unsigned short* __restrict__ Bt,
                            float* __restrict__ C,
                            int M, int N, int K, int addTo) {
  __shared__ unsigned short As[2][64 * 32];   // [m][k]  2 x 4 KB
  __shared__ unsigned short Bs[2][64 * 32];   // [n][k]  2 x 4 KB
  const int tid = threadIdx.x;
  const int wave = tid >> 5, lane = tid & 31;
  const int hi = lane >> 4, lr = lane & 15;
  const int tm = wave >> 2, tn = wave & 3;
  const long rowBase = (long)blockIdx.y * 64;
  const long colBase = (long)blockIdx.x * 64;

  FragF acc;
#pragma unroll
  for (int v = 0; v < 8; ++v) acc.f[v] = 0.f;

  // async-copy chunks: threads 0..255 -> A (256x16B), 256..511 -> Bt (256x16B)
  const int ta = tid & 255;
  const int row8 = ta >> 2, seg = ta & 3;     // 64 rows x 4 chunks of 16B
  const unsigned aL0 = lds_off(&As[0][row8 * 32 + seg * 8]);
  const unsigned aL1 = lds_off(&As[1][row8 * 32 + seg * 8]);
  const unsigned bL0 = lds_off(&Bs[0][row8 * 32 + seg * 8]);
  const unsigned bL1 = lds_off(&Bs[1][row8 * 32 + seg * 8]);
  const unsigned short* aG = A  + (size_t)(rowBase + row8) * K + seg * 8;
  const unsigned short* bG = Bt + (size_t)(colBase + row8) * K + seg * 8;

  const int arow = tm * 16 + lr;
  const int brow = tn * 16 + lr;

  // prologue: stage tile 0 into buffer 0
  if (tid < 256) async_copy16(aL0, aG);
  else           async_copy16(bL0, bG);

  int cur = 0;
  for (int kb = 0; kb < K; kb += 32, cur ^= 1) {
    const bool more = (kb + 32 < K);
    if (more) {
      if (tid < 256) async_copy16(cur ? aL0 : aL1, aG + kb + 32);
      else           async_copy16(cur ? bL0 : bL1, bG + kb + 32);
      __builtin_prefetch(aG + kb + 64, 0, 0);
      wait_async1();
    } else {
      wait_async0();
    }
    __syncthreads();

    FragBF a, b;
    const unsigned int* Au = (const unsigned int*)As[cur];
    const unsigned int* Bu = (const unsigned int*)Bs[cur];
#pragma unroll
    for (int v = 0; v < 8; ++v) {
      int ku = (v < 4 ? v : v + 4) + hi * 4;    // u32 (k-pair) index, ISA layout
      a.u[v] = Au[arow * 16 + ku];
      b.u[v] = Bu[brow * 16 + ku];
    }
    acc.v = __builtin_amdgcn_wmma_f32_16x16x32_bf16(false, a.v, false, b.v,
                                                    (short)0, acc.v, false, false);
    __syncthreads();
  }
  const long n = colBase + tn * 16 + lr;
#pragma unroll
  for (int v = 0; v < 8; ++v) {
    long m = rowBase + tm * 16 + v + 8 * hi;
    size_t off = (size_t)m * N + n;
    C[off] = addTo ? (C[off] + acc.f[v]) : acc.f[v];
  }
}

// Fused SwiGLU: Out[M,N](bf16) = silu(A@B1) * (A@B3); B1t/B3t are [N,K].
__global__ void k_gemm_swiglu(const unsigned short* __restrict__ A,
                              const unsigned short* __restrict__ B1t,
                              const unsigned short* __restrict__ B3t,
                              unsigned short* __restrict__ Out,
                              int M, int N, int K) {
  __shared__ unsigned short As[2][64 * 32];
  __shared__ unsigned short Bs1[2][64 * 32];
  __shared__ unsigned short Bs3[2][64 * 32];
  const int tid = threadIdx.x;
  const int wave = tid >> 5, lane = tid & 31;
  const int hi = lane >> 4, lr = lane & 15;
  const int tm = wave >> 2, tn = wave & 3;
  const long rowBase = (long)blockIdx.y * 64;
  const long colBase = (long)blockIdx.x * 64;

  FragF acc1, acc3;
#pragma unroll
  for (int v = 0; v < 8; ++v) { acc1.f[v] = 0.f; acc3.f[v] = 0.f; }

  const int ta = tid & 255;
  const int row8 = ta >> 2, seg = ta & 3;
  const unsigned aL[2]  = { lds_off(&As[0][row8 * 32 + seg * 8]),
                            lds_off(&As[1][row8 * 32 + seg * 8]) };
  const unsigned b1L[2] = { lds_off(&Bs1[0][row8 * 32 + seg * 8]),
                            lds_off(&Bs1[1][row8 * 32 + seg * 8]) };
  const unsigned b3L[2] = { lds_off(&Bs3[0][row8 * 32 + seg * 8]),
                            lds_off(&Bs3[1][row8 * 32 + seg * 8]) };
  const unsigned short* aG  = A   + (size_t)(rowBase + row8) * K + seg * 8;
  const unsigned short* b1G = B1t + (size_t)(colBase + row8) * K + seg * 8;
  const unsigned short* b3G = B3t + (size_t)(colBase + row8) * K + seg * 8;

  const int arow = tm * 16 + lr;
  const int brow = tn * 16 + lr;

  if (tid < 256) { async_copy16(aL[0], aG); async_copy16(b1L[0], b1G); }
  else           { async_copy16(b3L[0], b3G); }

  int cur = 0;
  for (int kb = 0; kb < K; kb += 32, cur ^= 1) {
    const bool more = (kb + 32 < K);
    if (more) {
      const int nxt = cur ^ 1;
      if (tid < 256) {
        async_copy16(aL[nxt], aG + kb + 32);
        async_copy16(b1L[nxt], b1G + kb + 32);
        wait_async2();
      } else {
        async_copy16(b3L[nxt], b3G + kb + 32);
        wait_async1();
      }
    } else {
      wait_async0();
    }
    __syncthreads();

    FragBF a, b1, b3;
    const unsigned int* Au  = (const unsigned int*)As[cur];
    const unsigned int* B1u = (const unsigned int*)Bs1[cur];
    const unsigned int* B3u = (const unsigned int*)Bs3[cur];
#pragma unroll
    for (int v = 0; v < 8; ++v) {
      int ku = (v < 4 ? v : v + 4) + hi * 4;
      a.u[v]  = Au[arow * 16 + ku];
      b1.u[v] = B1u[brow * 16 + ku];
      b3.u[v] = B3u[brow * 16 + ku];
    }
    acc1.v = __builtin_amdgcn_wmma_f32_16x16x32_bf16(false, a.v, false, b1.v,
                                                     (short)0, acc1.v, false, false);
    acc3.v = __builtin_amdgcn_wmma_f32_16x16x32_bf16(false, a.v, false, b3.v,
                                                     (short)0, acc3.v, false, false);
    __syncthreads();
  }
  const long n = colBase + tn * 16 + lr;
#pragma unroll
  for (int v = 0; v < 8; ++v) {
    long m = rowBase + tm * 16 + v + 8 * hi;
    float g = acc1.f[v];
    float val = (g / (1.f + __expf(-g))) * acc3.f[v];
    Out[(size_t)m * N + n] = f2bf(val);
  }
}

// ---------------------------------------------------------------- RoPE
__global__ void k_rope(float* __restrict__ Q, float* __restrict__ Kk,
                       const int* __restrict__ pos, const float* __restrict__ freqs) {
  size_t i = (size_t)blockIdx.x * blockDim.x + threadIdx.x;
  if (i >= (size_t)BB * SS * HH * (DHD / 2)) return;
  int d = (int)(i % (DHD / 2));
  size_t rem = i / (DHD / 2);
  int h = (int)(rem % HH);
  size_t bs = rem / HH;
  float ang = (float)pos[bs] * freqs[d];
  float c = cosf(ang), s = sinf(ang);
  size_t base = (bs * HH + h) * DHD + d;
  float q1 = Q[base], q2 = Q[base + DHD / 2];
  Q[base] = q1 * c - q2 * s;
  Q[base + DHD / 2] = q1 * s + q2 * c;
  float k1 = Kk[base], k2 = Kk[base + DHD / 2];
  Kk[base] = k1 * c - k2 * s;
  Kk[base + DHD / 2] = k1 * s + k2 * c;
}

// ---------------------------------------------------------------- attention
__global__ void k_attn(const float* __restrict__ Q, const float* __restrict__ Km,
                       const float* __restrict__ V, const unsigned char* __restrict__ mask,
                       float* __restrict__ O) {
  __shared__ float qs[WINS][DHD], ks[WINS][DHD], vs[WINS][DHD];
  __shared__ float sc[WINS][WINS];
  int blk = blockIdx.x;                     // b*NW*HH + w*HH + h
  int h = blk % HH;
  int w = (blk / HH) % NW;
  int b = blk / (HH * NW);
  int t = threadIdx.x;
  for (int e = t; e < WINS * DHD; e += 256) {
    int i = e / DHD, d = e % DHD;
    size_t base = ((size_t)(b * SS + w * WINS + i) * HH + h) * DHD + d;
    qs[i][d] = Q[base]; ks[i][d] = Km[base]; vs[i][d] = V[base];
  }
  __syncthreads();
  {
    int i = t >> 4, j = t & 15;
    float s = 0.f;
    for (int d = 0; d < DHD; ++d) s += qs[i][d] * ks[j][d];
    s *= 0.125f;   // 1/sqrt(64)
    if (!mask[b * SS + w * WINS + j]) s = -1e9f;
    sc[i][j] = s;
  }
  __syncthreads();
  if (t < WINS) {
    float mx = -1e30f;
    for (int j = 0; j < WINS; ++j) mx = fmaxf(mx, sc[t][j]);
    float sum = 0.f;
    for (int j = 0; j < WINS; ++j) { float e = __expf(sc[t][j] - mx); sc[t][j] = e; sum += e; }
    float inv = 1.f / sum;
    for (int j = 0; j < WINS; ++j) sc[t][j] *= inv;
  }
  __syncthreads();
  for (int e = t; e < WINS * DHD; e += 256) {
    int i = e / DHD, d = e % DHD;
    float o = 0.f;
    for (int j = 0; j < WINS; ++j) o += sc[i][j] * vs[j][d];
    O[((size_t)(b * SS + w * WINS + i) * HH + h) * DHD + d] = o;
  }
}

// ---------------------------------------------------------------- metric / merge
__global__ void k_real(const float* __restrict__ source, const float* __restrict__ pad,
                       int* __restrict__ realf) {
  __shared__ float red[256];
  int bs = blockIdx.x;
  int b = bs / SS;
  float acc = 0.f;
  for (int k2 = threadIdx.x; k2 < SS; k2 += 256)
    acc += source[(size_t)bs * SS + k2] * pad[b * SS + k2];
  red[threadIdx.x] = acc; __syncthreads();
  for (int s = 128; s > 0; s >>= 1) {
    if ((int)threadIdx.x < s) red[threadIdx.x] += red[threadIdx.x + s];
    __syncthreads();
  }
  if (threadIdx.x == 0) realf[bs] = red[0] > 0.f ? 1 : 0;
}

__global__ void k_sim(const float* __restrict__ metric, const int* __restrict__ realf,
                      float* __restrict__ sim) {
  int i = blockIdx.x * blockDim.x + threadIdx.x;
  if (i >= BB * NPAIR) return;
  int b = i / NPAIR, p = i % NPAIR;
  const float* me = metric + ((size_t)b * SS + 2 * p) * GMD;
  const float* mo = me + GMD;
  float ne = 0.f, no = 0.f, dt = 0.f;
  for (int g = 0; g < GMD; ++g) { ne += me[g] * me[g]; no += mo[g] * mo[g]; dt += me[g] * mo[g]; }
  float v = dt / ((sqrtf(ne) + 1e-6f) * (sqrtf(no) + 1e-6f));
  if (!(realf[b * SS + 2 * p] && realf[b * SS + 2 * p + 1])) v = -10000.f;
  sim[i] = v;
}

__global__ void k_topk(const float* __restrict__ sim, int* __restrict__ flag) {
  __shared__ float vals[NPAIR];
  __shared__ int   idxs[NPAIR];
  int b = blockIdx.x, t = threadIdx.x;      // 512 threads
  for (int e = t; e < NPAIR; e += 512) {
    vals[e] = sim[b * NPAIR + e]; idxs[e] = e; flag[b * NPAIR + e] = 0;
  }
  __syncthreads();
  for (int k = 2; k <= NPAIR; k <<= 1)
    for (int j = k >> 1; j > 0; j >>= 1) {
      for (int e = t; e < NPAIR; e += 512) {
        int l = e ^ j;
        if (l > e) {
          bool asc = ((e & k) == 0);
          bool sw = asc ? (vals[e] > vals[l]) : (vals[e] < vals[l]);
          if (sw) {
            float tv = vals[e]; vals[e] = vals[l]; vals[l] = tv;
            int ti = idxs[e]; idxs[e] = idxs[l]; idxs[l] = ti;
          }
        }
      }
      __syncthreads();
    }
  for (int e = t; e < RR; e += 512) flag[b * NPAIR + idxs[NPAIR - RR + e]] = 1;
}

__global__ void k_keep(const int* __restrict__ flag, int* __restrict__ keep) {
  int b = blockIdx.x;
  if (threadIdx.x != 0) return;
  int cnt = 0;
  for (int s = 0; s < SS; ++s) {
    int dr = (s & 1) && flag[b * NPAIR + (s >> 1)];
    if (!dr && cnt < SKEEP) keep[b * SKEEP + cnt++] = s;
  }
}

__global__ void k_gather_x(const float* __restrict__ X, const int* __restrict__ keep,
                           const int* __restrict__ flag, const int* __restrict__ span,
                           float* __restrict__ out) {
  int blk = blockIdx.x;
  int b = blk / SKEEP;
  int s = keep[blk];
  size_t srow = ((size_t)b * SS + s) * DD;
  size_t orow = (size_t)blk * DD;
  bool merge = ((s & 1) == 0) && flag[b * NPAIR + (s >> 1)];
  if (merge) {
    float sze = (float)span[b * SS + s], szo = (float)span[b * SS + s + 1];
    float inv = 1.f / (sze + szo);
    for (int d = threadIdx.x; d < DD; d += 256)
      out[orow + d] = (sze * X[srow + d] + szo * X[srow + DD + d]) * inv;
  } else {
    for (int d = threadIdx.x; d < DD; d += 256)
      out[orow + d] = X[srow + d];
  }
}

__global__ void k_gather_src(const float* __restrict__ source, const int* __restrict__ keep,
                             const int* __restrict__ flag, float* __restrict__ out) {
  int blk = blockIdx.x;
  int b = blk / SKEEP;
  int s = keep[blk];
  size_t srow = ((size_t)b * SS + s) * SS;
  size_t orow = (size_t)blk * SS;
  bool merge = ((s & 1) == 0) && flag[b * NPAIR + (s >> 1)];
  for (int k2 = threadIdx.x; k2 < SS; k2 += 256) {
    float v = source[srow + k2];
    if (merge) v += source[srow + SS + k2];
    out[orow + k2] = v;
  }
}

__global__ void k_gather_ps(const int* __restrict__ keep, const int* __restrict__ flag,
                            const int* __restrict__ pos, const int* __restrict__ span,
                            int* __restrict__ pout, int* __restrict__ spout) {
  int i = blockIdx.x * blockDim.x + threadIdx.x;
  if (i >= BB * SKEEP) return;
  int b = i / SKEEP;
  int s = keep[i];
  pout[i] = pos[b * SS + s];
  int sp = span[b * SS + s];
  if (((s & 1) == 0) && flag[b * NPAIR + (s >> 1)]) sp += span[b * SS + s + 1];
  spout[i] = sp;
}

// ---------------------------------------------------------------- launch
extern "C" void kernel_launch(void* const* d_in, const int* in_sizes, int n_in,
                              void* d_out, int out_size, void* d_ws, size_t ws_size,
                              hipStream_t stream) {
  (void)in_sizes; (void)n_in; (void)out_size; (void)ws_size;
  const float* in_x    = (const float*)d_in[0];
  const float* in_src  = (const float*)d_in[1];
  const int*   in_pos  = (const int*)d_in[2];
  const int*   in_span = (const int*)d_in[3];
  const float* in_freq = (const float*)d_in[5];
  const float* in_pad  = (const float*)d_in[6];
  const unsigned char* in_smask = (const unsigned char*)d_in[7];
  const float* in_wspan = (const float*)d_in[8];
  const float* in_wn1  = (const float*)d_in[9];
  const float* in_wq   = (const float*)d_in[10];
  const float* in_wk   = (const float*)d_in[11];
  const float* in_wv   = (const float*)d_in[12];
  const float* in_wo   = (const float*)d_in[13];
  const float* in_wn2  = (const float*)d_in[14];
  const float* in_w1   = (const float*)d_in[15];
  const float* in_w3   = (const float*)d_in[16];
  const float* in_w2   = (const float*)d_in[17];
  const float* in_wm   = (const float*)d_in[18];

  char* ws = (char*)d_ws;
  const size_t MB = 1u << 20;
  float* X  = (float*)(ws);                 // 32 MB running activations
  float* Q  = (float*)(ws + 32 * MB);       // 32 MB
  float* Kq = (float*)(ws + 64 * MB);       // 32 MB
  float* V  = (float*)(ws + 96 * MB);       // 32 MB
  float* O  = (float*)(ws + 128 * MB);      // 32 MB attention output
  unsigned short* Hb = (unsigned short*)(ws + 160 * MB); // 16 MB bf16 acts
  unsigned short* Wb = (unsigned short*)(ws + 176 * MB); // 16 MB bf16 weights (transposed)
  unsigned short* FFNact = (unsigned short*)(ws + 32 * MB); // overlays Q,K (64 MB)
  float* metric = (float*)(ws + 96 * MB);   // overlays V (2 MB)
  int*   realf  = (int*)(ws + 96 * MB + 2 * MB);
  float* sim    = (float*)((char*)realf + (size_t)BB * SS * 4);
  int*   flag   = (int*)((char*)sim + (size_t)BB * NPAIR * 4);
  int*   keep   = (int*)((char*)flag + (size_t)BB * NPAIR * 4);

  float* outf    = (float*)d_out;
  float* out_x   = outf;
  float* out_src = outf + (size_t)BB * SKEEP * DD;
  int*   out_pos = (int*)(outf + (size_t)BB * SKEEP * DD + (size_t)BB * SKEEP * SS);
  int*   out_sp  = out_pos + BB * SKEEP;

  const size_t NTOK = (size_t)BB * SS;           // 8192
  const size_t NELT = NTOK * DD;                 // 8388608

  // 1) x += span embedding
  k_span_add<<<(unsigned)((NELT + 255) / 256), 256, 0, stream>>>(in_x, in_span, in_smask, in_wspan, X);
  // 2) RMSNorm1 -> bf16
  k_rms_bf16<<<(unsigned)NTOK, 256, 0, stream>>>(X, in_wn1, Hb);
  // 3) transpose-cast wq/wk/wv, QKV GEMMs
  dim3 tQ(DD / 32, DD / 32);
  k_castT_bf16<<<tQ, 256, 0, stream>>>(in_wq, Wb, DD, DD);
  k_castT_bf16<<<tQ, 256, 0, stream>>>(in_wk, Wb + (size_t)DD * DD, DD, DD);
  k_castT_bf16<<<tQ, 256, 0, stream>>>(in_wv, Wb + 2 * (size_t)DD * DD, DD, DD);
  dim3 gq(DD / 64, (unsigned)(NTOK / 64));
  k_gemm_bf16<<<gq, 512, 0, stream>>>(Hb, Wb,                       Q,  (int)NTOK, DD, DD, 0);
  k_gemm_bf16<<<gq, 512, 0, stream>>>(Hb, Wb + (size_t)DD * DD,     Kq, (int)NTOK, DD, DD, 0);
  k_gemm_bf16<<<gq, 512, 0, stream>>>(Hb, Wb + 2 * (size_t)DD * DD, V,  (int)NTOK, DD, DD, 0);
  // 4) RoPE on Q,K
  size_t nrope = NTOK * HH * (DHD / 2);
  k_rope<<<(unsigned)((nrope + 255) / 256), 256, 0, stream>>>(Q, Kq, in_pos, in_freq);
  // 5) windowed attention
  k_attn<<<BB * NW * HH, 256, 0, stream>>>(Q, Kq, V, in_smask, O);
  // 6) x += O @ wo
  k_cast_bf16<<<(unsigned)((NELT + 255) / 256), 256, 0, stream>>>(O, Hb, (int)NELT);
  k_castT_bf16<<<tQ, 256, 0, stream>>>(in_wo, Wb, DD, DD);
  k_gemm_bf16<<<gq, 512, 0, stream>>>(Hb, Wb, X, (int)NTOK, DD, DD, 1);
  // 7) RMSNorm2 -> bf16
  k_rms_bf16<<<(unsigned)NTOK, 256, 0, stream>>>(X, in_wn2, Hb);
  // 8) SwiGLU activation (fused dual GEMM) -> bf16
  dim3 tF(FFD / 32, DD / 32);
  k_castT_bf16<<<tF, 256, 0, stream>>>(in_w1, Wb, DD, FFD);
  k_castT_bf16<<<tF, 256, 0, stream>>>(in_w3, Wb + (size_t)DD * FFD, DD, FFD);
  dim3 gf(FFD / 64, (unsigned)(NTOK / 64));
  k_gemm_swiglu<<<gf, 512, 0, stream>>>(Hb, Wb, Wb + (size_t)DD * FFD, FFNact, (int)NTOK, FFD, DD);
  // 9) x += act @ w2
  dim3 tW2(DD / 32, FFD / 32);
  k_castT_bf16<<<tW2, 256, 0, stream>>>(in_w2, Wb, FFD, DD);
  k_gemm_bf16<<<gq, 512, 0, stream>>>(FFNact, Wb, X, (int)NTOK, DD, FFD, 1);
  // 10) metric = x @ w_metric
  k_cast_bf16<<<(unsigned)((NELT + 255) / 256), 256, 0, stream>>>(X, Hb, (int)NELT);
  dim3 tM(GMD / 32, DD / 32);
  k_castT_bf16<<<tM, 256, 0, stream>>>(in_wm, Wb, DD, GMD);
  dim3 gm(GMD / 64, (unsigned)(NTOK / 64));
  k_gemm_bf16<<<gm, 512, 0, stream>>>(Hb, Wb, metric, (int)NTOK, GMD, DD, 0);
  // 11) pair similarity + top-k merge selection
  k_real<<<(unsigned)NTOK, 256, 0, stream>>>(in_src, in_pad, realf);
  k_sim<<<(BB * NPAIR + 255) / 256, 256, 0, stream>>>(metric, realf, sim);
  k_topk<<<BB, 512, 0, stream>>>(sim, flag);
  k_keep<<<BB, 32, 0, stream>>>(flag, keep);
  // 12) gathers
  k_gather_x<<<BB * SKEEP, 256, 0, stream>>>(X, keep, flag, in_span, out_x);
  k_gather_src<<<BB * SKEEP, 256, 0, stream>>>(in_src, keep, flag, out_src);
  k_gather_ps<<<(BB * SKEEP + 255) / 256, 256, 0, stream>>>(keep, flag, in_pos, in_span, out_pos, out_sp);
}
